// SwinTransformerCustom_88562225644244
// MI455X (gfx1250) — matrix operand order repaired and checked by reading, not compiled, not run on hardware
//
// Swin-Transformer forward for MI455X (gfx1250), wave32 + WMMA f16->f32.
// All GEMMs (qkv / proj / fc1+gelu / fc2 / patch-merge / conv head) and the
// windowed attention (QK^T and P*V) run through v_wmma_f32_16x16x32_f16.
// Tile staging uses gfx1250 async global->LDS loads when the builtins exist.
#include <hip/hip_runtime.h>
#include <hip/hip_bf16.h>

typedef _Float16 f16;
typedef __attribute__((ext_vector_type(16))) _Float16 v16h;
typedef __attribute__((ext_vector_type(8)))  float    v8f;
typedef __attribute__((ext_vector_type(4)))  int      i32x4;

union Frag16 { v16h v; f16 h[16]; unsigned u[8]; };
union Acc8   { v8f  v; float f[8]; };

// gfx1250 async global->LDS copy path (compile-safe feature detection)
#if defined(__has_builtin)
#  if __has_builtin(__builtin_amdgcn_global_load_async_to_lds_b128) && \
      __has_builtin(__builtin_amdgcn_s_wait_asynccnt)
#    define GFX1250_ASYNC_LDS 1
#  endif
#endif
#ifndef GFX1250_ASYNC_LDS
#  define GFX1250_ASYNC_LDS 0
#endif

// address-space-qualified int4 for the async-LDS builtins
typedef __attribute__((address_space(1))) i32x4 g_i32x4;   // global
typedef __attribute__((address_space(3))) i32x4 l_i32x4;   // LDS

__device__ __forceinline__ float gelu_exact(float x) {
  return 0.5f * x * (1.0f + erff(x * 0.70710678118654752f));
}

// ---------------------------------------------------------------- utilities
__global__ void f32_to_f16_kernel(const float* __restrict__ in, f16* __restrict__ out, int n) {
  int i = blockIdx.x * 256 + threadIdx.x;
  if (i < n) out[i] = (f16)in[i];
}

__global__ void expand_bias_kernel(const float* __restrict__ b, float* __restrict__ out, int n, int rep) {
  int i = blockIdx.x * 256 + threadIdx.x;
  if (i < n) out[i] = b[i / rep];
}

// one wave32 per row layernorm; writes optional f16 and/or f32 outputs
__global__ void layernorm_kernel(const float* __restrict__ x, const float* __restrict__ w,
                                 const float* __restrict__ b, f16* __restrict__ outh,
                                 float* __restrict__ outf, int M, int C) {
  int row  = blockIdx.x * 8 + (threadIdx.x >> 5);
  int lane = threadIdx.x & 31;
  if (row >= M) return;
  const float* xr = x + (size_t)row * C;
  float s = 0.f, s2 = 0.f;
  for (int c = lane; c < C; c += 32) { float v = xr[c]; s += v; s2 += v * v; }
  for (int off = 16; off; off >>= 1) { s += __shfl_xor(s, off, 32); s2 += __shfl_xor(s2, off, 32); }
  float mean = s / C;
  float var  = s2 / C - mean * mean;
  float inv  = rsqrtf(var + 1e-5f);
  for (int c = lane; c < C; c += 32) {
    float v = (xr[c] - mean) * inv * w[c] + b[c];
    if (outh) outh[(size_t)row * C + c] = (f16)v;
    if (outf) outf[(size_t)row * C + c] = v;
  }
}

// patch embed: out[b,h,w,o] = sum_{c,p,q} x[b,c,4h+p,4w+q]*w[o,c,p,q] + b[o]
__global__ void patch_embed_kernel(const float* __restrict__ x, const float* __restrict__ w,
                                   const float* __restrict__ b, float* __restrict__ out) {
  int tok = blockIdx.x;                 // b*256 + h*16 + w
  int bb = tok >> 8; int hw = tok & 255; int h = hw >> 4; int wc = hw & 15;
  __shared__ float patch[48];
  int t = threadIdx.x;                  // 96 threads, one output channel each
  if (t < 48) {
    int c = t >> 4, p = (t >> 2) & 3, q = t & 3;
    patch[t] = x[(((size_t)bb * 3 + c) * 64 + (h * 4 + p)) * 64 + (wc * 4 + q)];
  }
  __syncthreads();
  float acc = b[t];
  const float* wr = w + t * 48;
  #pragma unroll 8
  for (int k = 0; k < 48; ++k) acc += patch[k] * wr[k];
  out[(size_t)tok * 96 + t] = acc;
}

// gather 2x2 neighborhoods into 4C rows (patch merging input)
__global__ void merge_gather_kernel(const float* __restrict__ h, float* __restrict__ out,
                                    int Bimg, int res, int C) {
  int res2 = res >> 1;
  size_t total = (size_t)Bimg * res2 * res2 * 4 * C;
  for (size_t i = blockIdx.x * 256ull + threadIdx.x; i < total; i += (size_t)gridDim.x * 256) {
    int c = (int)(i % (4 * C)); size_t t = i / (4 * C);
    int w2 = (int)(t % res2); size_t t2 = t / res2;
    int h2 = (int)(t2 % res2); int b = (int)(t2 / res2);
    int part = c / C; int cc = c % C;
    int dh = (part == 1 || part == 3) ? 1 : 0;
    int dw = (part >= 2) ? 1 : 0;
    out[i] = h[(((size_t)b * res + (2 * h2 + dh)) * res + (2 * w2 + dw)) * C + cc];
  }
}

// ------------------------------------------------------------- WMMA GEMM
// D[M,N] = A[M,K](f16) x W^T (+bias, +gelu, +fp32 residual)
// TRANSB=true : W stored (N,K)   (all nn.Linear weights)
// TRANSB=false: W stored (K,N)   (final conv head)
#define BM 128
#define BN 64
template <bool TRANSB>
__global__ void __launch_bounds__(256)
gemm_wmma_kernel(const f16* __restrict__ A, const f16* __restrict__ W,
                 const float* __restrict__ bias, const float* __restrict__ resid,
                 f16* __restrict__ outh, float* __restrict__ outf,
                 int M, int N, int K, int do_gelu) {
  __shared__ __align__(16) f16 a_lds[BM][40];
  __shared__ __align__(16) f16 b_lds[BN][40];

  int tid = threadIdx.x;
  int wave = tid >> 5, lane = tid & 31;
  int wm = wave >> 1, wn = wave & 1;          // 4 x 2 wave grid -> 32x32 per wave
  int m0 = blockIdx.y * BM;
  int n0 = blockIdx.x * BN;
  int half = lane >> 4, lr = lane & 15;

  const bool afull = (m0 + BM <= M);          // true for every launch here (M % 128 == 0)
  const bool bfull = (n0 + BN <= N);

  Acc8 acc[2][2];
  #pragma unroll
  for (int mi = 0; mi < 2; ++mi)
    #pragma unroll
    for (int ni = 0; ni < 2; ++ni)
      #pragma unroll
      for (int r = 0; r < 8; ++r) acc[mi][ni].f[r] = 0.f;

  for (int k0 = 0; k0 < K; k0 += 32) {
    __syncthreads();
    // ---- A tile: BM x 32 halves, 128-bit transfers (2 per thread)
    if (afull) {
      #pragma unroll
      for (int e = 0; e < 2; ++e) {
        int el = tid + e * 256;               // 0..511 : 128 rows x 4 chunks
        int m = el >> 2, ch = (el & 3) * 8;
        const f16* src = A + (size_t)(m0 + m) * K + k0 + ch;
#if GFX1250_ASYNC_LDS
        __builtin_amdgcn_global_load_async_to_lds_b128(
            (g_i32x4*)src, (l_i32x4*)&a_lds[m][ch], 0, 0);
#else
        *(uint4*)&a_lds[m][ch] = *(const uint4*)src;
#endif
      }
    } else {
      for (int i = tid; i < BM * 16; i += 256) {
        int m = i >> 4; int kk = (i & 15) * 2;
        int gm = m0 + m;
        unsigned val = 0;
        if (gm < M) val = *(const unsigned*)(A + (size_t)gm * K + k0 + kk);
        *(unsigned*)&a_lds[m][kk] = val;
      }
    }
    // ---- B tile
    if (TRANSB) {
      if (bfull) {
        int n = tid >> 2, ch = (tid & 3) * 8; // 64 rows x 4 chunks = 256
        const f16* src = W + (size_t)(n0 + n) * K + k0 + ch;
#if GFX1250_ASYNC_LDS
        __builtin_amdgcn_global_load_async_to_lds_b128(
            (g_i32x4*)src, (l_i32x4*)&b_lds[n][ch], 0, 0);
#else
        *(uint4*)&b_lds[n][ch] = *(const uint4*)src;
#endif
      } else {
        for (int i = tid; i < BN * 16; i += 256) {
          int n = i >> 4; int kk = (i & 15) * 2;
          int gn = n0 + n;
          unsigned val = 0;
          if (gn < N) val = *(const unsigned*)(W + (size_t)gn * K + k0 + kk);
          *(unsigned*)&b_lds[n][kk] = val;
        }
      }
    } else {
      for (int i = tid; i < BN * 32; i += 256) {
        int n = i >> 5; int kk = i & 31;
        int gn = n0 + n;
        f16 val = (f16)0.f;
        if (gn < N) val = W[(size_t)(k0 + kk) * N + gn];
        b_lds[n][kk] = val;
      }
    }
#if GFX1250_ASYNC_LDS
    __builtin_amdgcn_s_wait_asynccnt(0);
#endif
    __syncthreads();

    Frag16 af[2], bf[2];
    #pragma unroll
    for (int mi = 0; mi < 2; ++mi) {
      int mrow = wm * 32 + mi * 16 + lr;
      #pragma unroll
      for (int v = 0; v < 8; ++v) {
        int kb = ((v >= 4) ? 16 : 0) + half * 8 + (v & 3) * 2;
        af[mi].u[v] = *(const unsigned*)&a_lds[mrow][kb];
      }
    }
    #pragma unroll
    for (int ni = 0; ni < 2; ++ni) {
      int ncol = wn * 32 + ni * 16 + lr;
      #pragma unroll
      for (int v = 0; v < 8; ++v) {
        int kb = ((v >= 4) ? 16 : 0) + half * 8 + (v & 3) * 2;
        bf[ni].u[v] = *(const unsigned*)&b_lds[ncol][kb];
      }
    }
    #pragma unroll
    for (int mi = 0; mi < 2; ++mi)
      #pragma unroll
      for (int ni = 0; ni < 2; ++ni)
        acc[mi][ni].v = __builtin_amdgcn_wmma_f32_16x16x32_f16(
            false, af[mi].v, false, bf[ni].v, (short)0, acc[mi][ni].v, false, false);
  }

  #pragma unroll
  for (int mi = 0; mi < 2; ++mi)
    #pragma unroll
    for (int ni = 0; ni < 2; ++ni) {
      int col = n0 + wn * 32 + ni * 16 + lr;
      if (col >= N) continue;
      #pragma unroll
      for (int r = 0; r < 8; ++r) {
        int row = m0 + wm * 32 + mi * 16 + r + half * 8;
        if (row >= M) continue;
        float v = acc[mi][ni].f[r];
        if (bias)   v += bias[col];
        if (do_gelu) v = gelu_exact(v);
        size_t idx = (size_t)row * N + col;
        if (resid) v += resid[idx];
        if (outh) outh[idx] = (f16)v;
        if (outf) outf[idx] = v;
      }
    }
}

// ------------------------------------------------------------- attention
// One block per (window, head). NP = padded window tokens (64 for N=49, 16 for N<=16).
template <int NP, int NTHREADS>
__global__ void __launch_bounds__(NTHREADS)
attn_kernel(const f16* __restrict__ qkv, f16* __restrict__ attn_out,
            const float* __restrict__ rpb,
            int res, int C, int nh, int ws, int ss, int N, int nWinSide, float scale) {
  constexpr int KP = (NP < 32) ? 32 : NP;     // K-extent for P*V (zero padded)
  constexpr int NWv = NP / 16;
  constexpr int NWAVES = NTHREADS / 32;

  __shared__ __align__(16) f16  qs[NP][40];
  __shared__ __align__(16) f16  ks[NP][40];
  __shared__ __align__(16) f16  vt[32][KP + 8];   // transposed V: vt[d][token]
  __shared__ __align__(16) float sc[NP][NP + 1];
  __shared__ __align__(16) f16  ps[NP][KP + 8];
  __shared__ int tokidx[NP];
  __shared__ int reg[NP];

  int bid = blockIdx.x;
  int hh  = bid % nh;
  int wid = bid / nh;
  int nW2 = nWinSide * nWinSide;
  int img = wid / nW2; int wrem = wid % nW2;
  int whi = wrem / nWinSide; int wwi = wrem % nWinSide;
  int tid = threadIdx.x;
  int C3 = 3 * C;

  for (int t = tid; t < NP; t += NTHREADS) {
    int tok = -1, rg = 0;
    if (t < N) {
      int th = t / ws, tw = t % ws;
      int hr = whi * ws + th, wr = wwi * ws + tw;   // coords in rolled+padded image
      int Hp = nWinSide * ws;
      if (ss > 0) {
        int rh = (hr < Hp - ws) ? 0 : ((hr < Hp - ss) ? 1 : 2);
        int rw = (wr < Hp - ws) ? 0 : ((wr < Hp - ss) ? 1 : 2);
        rg = rh * 3 + rw;
      }
      if (hr < res && wr < res) {
        int hs = (hr + ss) % res; int wsc = (wr + ss) % res;
        tok = (img * res + hs) * res + wsc;
      }
    }
    tokidx[t] = tok; reg[t] = rg;
  }
  __syncthreads();

  for (int i = tid; i < KP * 32; i += NTHREADS) {
    int t = i >> 5, d = i & 31;
    f16 qv = (f16)0.f, kv = (f16)0.f, vv = (f16)0.f;
    int tok = (t < NP) ? tokidx[t] : -1;
    if (tok >= 0) {
      size_t base = (size_t)tok * C3 + hh * 32 + d;
      qv = qkv[base]; kv = qkv[base + C]; vv = qkv[base + 2 * C];
    }
    if (t < NP) { qs[t][d] = qv; ks[t][d] = kv; }
    vt[d][t] = vv;
  }
  __syncthreads();

  int wave = tid >> 5, lane = tid & 31;
  int half = lane >> 4, lr = lane & 15;

  // S = q*k^T (head dim 32 -> one WMMA per 16x16 tile), + bias + masks
  for (int tile = wave; tile < NWv * NWv; tile += NWAVES) {
    int ti = tile / NWv, tj = tile % NWv;
    Frag16 af, bf; Acc8 acc;
    #pragma unroll
    for (int r = 0; r < 8; ++r) acc.f[r] = 0.f;
    #pragma unroll
    for (int v = 0; v < 8; ++v) {
      int kba = ((v >= 4) ? 16 : 0) + half * 8 + (v & 3) * 2;
      af.u[v] = *(const unsigned*)&qs[ti * 16 + lr][kba];
      int kbb = ((v >= 4) ? 16 : 0) + half * 8 + (v & 3) * 2;
      bf.u[v] = *(const unsigned*)&ks[tj * 16 + lr][kbb];
    }
    acc.v = __builtin_amdgcn_wmma_f32_16x16x32_f16(false, af.v, false, bf.v,
                                                   (short)0, acc.v, false, false);
    #pragma unroll
    for (int r = 0; r < 8; ++r) {
      int i = ti * 16 + r + half * 8;
      int j = tj * 16 + lr;
      float val = acc.f[r] * scale;
      if (i < N && j < N) {
        int ih = i / ws, iw = i % ws, jh = j / ws, jw = j % ws;
        int rel = (ih - jh + ws - 1) * (2 * ws - 1) + (iw - jw + ws - 1);
        val += rpb[rel * nh + hh];
        if (ss > 0 && reg[i] != reg[j]) val += -100.f;
      }
      if (j >= N) val = -1e30f;
      sc[i][j] = val;
    }
  }
  __syncthreads();

  // softmax per row, write f16 P (zero-padded to KP columns)
  for (int i = tid; i < NP; i += NTHREADS) {
    float mx = -1e30f;
    for (int j = 0; j < NP; ++j) mx = fmaxf(mx, sc[i][j]);
    float sum = 0.f;
    for (int j = 0; j < NP; ++j) sum += __expf(sc[i][j] - mx);
    float inv = (sum > 0.f) ? 1.f / sum : 0.f;
    for (int j = 0; j < KP; ++j) {
      float p = (j < NP) ? __expf(sc[i][j] - mx) * inv : 0.f;
      ps[i][j] = (f16)p;
    }
  }
  __syncthreads();

  // out = P @ V : NP x 32, K = KP
  for (int tile = wave; tile < NWv * 2; tile += NWAVES) {
    int om = tile >> 1, on = tile & 1;
    Acc8 acc;
    #pragma unroll
    for (int r = 0; r < 8; ++r) acc.f[r] = 0.f;
    for (int k0 = 0; k0 < KP; k0 += 32) {
      Frag16 af, bf;
      #pragma unroll
      for (int v = 0; v < 8; ++v) {
        int kba = k0 + ((v >= 4) ? 16 : 0) + half * 8 + (v & 3) * 2;
        af.u[v] = *(const unsigned*)&ps[om * 16 + lr][kba];
        int kbb = k0 + ((v >= 4) ? 16 : 0) + half * 8 + (v & 3) * 2;
        bf.u[v] = *(const unsigned*)&vt[on * 16 + lr][kbb];
      }
      acc.v = __builtin_amdgcn_wmma_f32_16x16x32_f16(false, af.v, false, bf.v,
                                                     (short)0, acc.v, false, false);
    }
    #pragma unroll
    for (int r = 0; r < 8; ++r) {
      int i = om * 16 + r + half * 8;
      int d = on * 16 + lr;
      if (i < N) {
        int tok = tokidx[i];
        if (tok >= 0) attn_out[(size_t)tok * C + hh * 32 + d] = (f16)acc.f[r];
      }
    }
  }
}

// ------------------------------------------------------------- host driver
static void conv16(const float* src, f16* dst, int n, hipStream_t s) {
  f32_to_f16_kernel<<<(n + 255) / 256, 256, 0, s>>>(src, dst, n);
}
static void gemm(const f16* A, const f16* W, const float* bias, const float* resid,
                 f16* outh, float* outf, int M, int N, int K, int do_gelu, bool transb,
                 hipStream_t s) {
  dim3 g((N + BN - 1) / BN, (M + BM - 1) / BM);
  if (transb) gemm_wmma_kernel<true ><<<g, 256, 0, s>>>(A, W, bias, resid, outh, outf, M, N, K, do_gelu);
  else        gemm_wmma_kernel<false><<<g, 256, 0, s>>>(A, W, bias, resid, outh, outf, M, N, K, do_gelu);
}

extern "C" void kernel_launch(void* const* d_in, const int* in_sizes, int n_in,
                              void* d_out, int out_size, void* d_ws, size_t ws_size,
                              hipStream_t stream) {
  (void)in_sizes; (void)n_in; (void)out_size; (void)ws_size;
  static const int kDepth[4] = {2, 2, 6, 2};
  static const int kHeads[4] = {3, 6, 12, 24};
  static const int kDims [4] = {96, 192, 384, 768};

  // ---- decode inputs: 'x' then params in jax-pytree (sorted key) order
  int idx = 0;
  auto P = [&]() { return (const float*)d_in[idx++]; };
  const float* x_in   = P();
  const float* conv_b = P();
  const float* conv_w = P();   // (3072, 128*8*8) row-major
  const float* ln_nb  = P();
  const float* ln_nw  = P();
  const float* pe_b   = P();
  const float* pe_nb  = P();
  const float* pe_nw  = P();
  const float* pe_w   = P();   // (96, 48)
  struct BW { const float *fc1_b,*fc1_w,*fc2_b,*fc2_w,*n1b,*n1w,*n2b,*n2w,*proj_b,*proj_w,*qkv_b,*qkv_w,*rpb; };
  BW blk[12]; const float* mg_nb[3]; const float* mg_nw[3]; const float* mg_rw[3];
  int bi = 0;
  for (int s = 0; s < 4; ++s) {
    for (int d = 0; d < kDepth[s]; ++d) {
      BW& b = blk[bi++];
      b.fc1_b = P(); b.fc1_w = P(); b.fc2_b = P(); b.fc2_w = P();
      b.n1b = P(); b.n1w = P(); b.n2b = P(); b.n2w = P();
      b.proj_b = P(); b.proj_w = P(); b.qkv_b = P(); b.qkv_w = P(); b.rpb = P();
    }
    if (s < 3) { mg_nb[s] = P(); mg_nw[s] = P(); mg_rw[s] = P(); }
  }

  // ---- workspace layout (256B aligned)
  char* wp = (char*)d_ws;
  auto wsalloc = [&](size_t bytes) { void* p = wp; wp += (bytes + 255) & ~(size_t)255; return p; };
  const size_t M0 = 65536;  // 256 * 16 * 16
  float* hbuf  = (float*)wsalloc(M0 * 96 * 4);        // residual stream (fp32)
  float* tmpf  = (float*)wsalloc(M0 * 96 * 4);        // patch-embed raw / merge gather
  f16*  ln16   = (f16*) wsalloc(M0 * 96 * 2);
  f16*  qkv16  = (f16*) wsalloc(M0 * 288 * 2);
  f16*  attn16 = (f16*) wsalloc(M0 * 96 * 2);
  f16*  hid16  = (f16*) wsalloc(M0 * 384 * 2);
  f16*  mrg16  = (f16*) wsalloc((M0 / 4) * 384 * 2);
  f16*  wbuf   = (f16*) wsalloc((size_t)3072 * 8192 * 2);
  float* bexp  = (float*)wsalloc(8192 * 4);

  // ---- patch embed + LN -> hbuf
  patch_embed_kernel<<<(int)M0, 96, 0, stream>>>(x_in, pe_w, pe_b, tmpf);
  layernorm_kernel<<<(int)((M0 + 7) / 8), 256, 0, stream>>>(tmpf, pe_nw, pe_nb, nullptr, hbuf, (int)M0, 96);

  // ---- stages
  const float scale = 0.17677669529663687f;  // 32^-0.5
  int res = 16; bi = 0;
  for (int s = 0; s < 4; ++s) {
    int C = kDims[s], nh = kHeads[s];
    int wsz = res < 7 ? res : 7;
    int N = wsz * wsz;
    bool shiftable = res > wsz;
    int nWinSide = (res + wsz - 1) / wsz;
    int M = 256 * res * res;
    int numWin = 256 * nWinSide * nWinSide;
    for (int d = 0; d < kDepth[s]; ++d) {
      const BW& b = blk[bi++];
      int ss = (shiftable && (d & 1)) ? (wsz / 2) : 0;
      // ln1
      layernorm_kernel<<<(M + 7) / 8, 256, 0, stream>>>(hbuf, b.n1w, b.n1b, ln16, nullptr, M, C);
      // qkv
      conv16(b.qkv_w, wbuf, 3 * C * C, stream);
      gemm(ln16, wbuf, b.qkv_b, nullptr, qkv16, nullptr, M, 3 * C, C, 0, true, stream);
      // windowed attention
      int grid = numWin * nh;
      if (N == 49)
        attn_kernel<64, 256><<<grid, 256, 0, stream>>>(qkv16, attn16, b.rpb, res, C, nh, wsz, ss, N, nWinSide, scale);
      else
        attn_kernel<16, 64 ><<<grid, 64, 0, stream>>>(qkv16, attn16, b.rpb, res, C, nh, wsz, ss, N, nWinSide, scale);
      // proj + residual (fp32 stream update)
      conv16(b.proj_w, wbuf, C * C, stream);
      gemm(attn16, wbuf, b.proj_b, hbuf, nullptr, hbuf, M, C, C, 0, true, stream);
      // ln2
      layernorm_kernel<<<(M + 7) / 8, 256, 0, stream>>>(hbuf, b.n2w, b.n2b, ln16, nullptr, M, C);
      // fc1 + exact gelu
      conv16(b.fc1_w, wbuf, 4 * C * C, stream);
      gemm(ln16, wbuf, b.fc1_b, nullptr, hid16, nullptr, M, 4 * C, C, 1, true, stream);
      // fc2 + residual
      conv16(b.fc2_w, wbuf, 4 * C * C, stream);
      gemm(hid16, wbuf, b.fc2_b, hbuf, nullptr, hbuf, M, C, 4 * C, 0, true, stream);
    }
    if (s < 3) {
      int M2 = M / 4, C4 = 4 * C;
      merge_gather_kernel<<<1024, 256, 0, stream>>>(hbuf, tmpf, 256, res, C);
      layernorm_kernel<<<(M2 + 7) / 8, 256, 0, stream>>>(tmpf, mg_nw[s], mg_nb[s], mrg16, nullptr, M2, C4);
      conv16(mg_rw[s], wbuf, 2 * C * C4, stream);
      gemm(mrg16, wbuf, nullptr, nullptr, nullptr, hbuf, M2, 2 * C, C4, 0, true, stream);
      res >>= 1;
    }
  }

  // ---- head: LN(768) over 1024 tokens -> feat (256,3072) @ conv_w (3072,8192) + bias
  layernorm_kernel<<<(1024 + 7) / 8, 256, 0, stream>>>(hbuf, ln_nw, ln_nb, ln16, nullptr, 1024, 768);
  conv16(conv_w, wbuf, 3072 * 8192, stream);
  expand_bias_kernel<<<32, 256, 0, stream>>>(conv_b, bexp, 8192, 64);
  gemm(ln16, wbuf, bexp, nullptr, nullptr, (float*)d_out, 256, 8192, 3072, 0, false, stream);
}
